// EncoderBlock_45509473468812
// MI455X (gfx1250) — matrix operand order repaired and checked by reading, not compiled, not run on hardware
//
#include <hip/hip_runtime.h>
#include <hip/hip_bf16.h>

typedef __attribute__((ext_vector_type(16))) _Float16 v16h;
typedef __attribute__((ext_vector_type(8)))  _Float16 v8h;
typedef __attribute__((ext_vector_type(8)))  float    v8f;

#define HID 64
#define LN_EPS 1e-5f

__device__ __forceinline__ v8f wmma_f16(v16h a, v16h b, v8f c) {
  // (neg_a, A, neg_b, B, c_mod, C, reuse_a, reuse_b)
  return __builtin_amdgcn_wmma_f32_16x16x32_f16(false, a, false, b, (short)0, c, false, false);
}

__device__ __forceinline__ v16h load_bfrag(const _Float16* frags, int fragIdx, int lane) {
  const v16h* p = (const v16h*)frags;   // each v16h = one lane's 32B slice of a fragment
  return p[fragIdx * 32 + lane];
}

// Pre-swizzle a [Kdim x 64] f32 weight matrix into WMMA B-fragment order (f16).
// Fragment(kc,ct): lane<16 -> N=lane,    K = kc*32 + (e>>3)*16 + (e&7)      (e=0..15)
//                  lane>=16 -> N=lane-16, K = kc*32 + (e>>3)*16 + 8 + (e&7)
__global__ void pack_weights_kernel(const float* __restrict__ W, _Float16* __restrict__ out, int nkc) {
  int idx = blockIdx.x * blockDim.x + threadIdx.x;
  int total = nkc * 2048;                 // nkc * 4 ct * 32 lanes * 16 elems
  if (idx >= total) return;
  int e    = idx & 15;
  int lane = (idx >> 4) & 31;
  int ct   = (idx >> 9) & 3;
  int kc   = idx >> 11;
  int k    = kc * 32 + ((e >> 3) * 16) + ((lane >> 4) * 8) + (e & 7);
  int col  = ct * 16 + (lane & 15);
  out[idx] = (_Float16)W[k * HID + col];
}

// ReLU(acc + bias) -> LDS staging (row-major 16x64 f16) -> A fragments -> next layer (K=64).
__device__ void relu_stage_next(v8f* acc, const float* __restrict__ bias, _Float16* st,
                                const _Float16* __restrict__ frags, int fragBase, int lane) {
  int half = lane >> 4, ln = lane & 15;
  #pragma unroll
  for (int ct = 0; ct < 4; ++ct) {
    float bb = bias[ct * 16 + ln];
    #pragma unroll
    for (int r = 0; r < 8; ++r) {
      float v = acc[ct][r] + bb;
      v = v > 0.f ? v : 0.f;
      st[(half * 8 + r) * HID + ct * 16 + ln] = (_Float16)v;
    }
  }
  __syncthreads();
  v16h a[2];
  #pragma unroll
  for (int kc = 0; kc < 2; ++kc) {
    // A layout: M = lane&15; elems 0..7 = K kc*32 + (half?8:0)+j ; elems 8..15 = +16
    const v8h* lo = (const v8h*)(st + ln * HID + kc * 32 + half * 8);
    const v8h* hi = (const v8h*)(st + ln * HID + kc * 32 + 16 + half * 8);
    v8h l = *lo, h = *hi;
    #pragma unroll
    for (int j = 0; j < 8; ++j) { a[kc][j] = l[j]; a[kc][8 + j] = h[j]; }
  }
  __syncthreads();
  #pragma unroll
  for (int ct = 0; ct < 4; ++ct) {
    v8f c = {0.f, 0.f, 0.f, 0.f, 0.f, 0.f, 0.f, 0.f};
    c = wmma_f16(a[0], load_bfrag(frags, fragBase + ct, lane), c);
    c = wmma_f16(a[1], load_bfrag(frags, fragBase + 4 + ct, lane), c);
    acc[ct] = c;
  }
}

// Edge MLP + LayerNorm + fused double scatter-add. One wave = 16 edge rows.
__global__ void __launch_bounds__(256)
edge_kernel(const float* __restrict__ xe, const _Float16* __restrict__ frags,
            const float* __restrict__ b0, const float* __restrict__ b1,
            const float* __restrict__ b2, const float* __restrict__ b3,
            const float* __restrict__ lns, const float* __restrict__ lnb,
            const int* __restrict__ eidx,
            float* __restrict__ out_edge, float* __restrict__ agg, long long E) {
  __shared__ __align__(16) _Float16 stage[8 * 16 * HID];
  int lane = threadIdx.x & 31, wave = threadIdx.x >> 5;
  int half = lane >> 4, ln = lane & 15;
  _Float16* st = &stage[wave * 16 * HID];
  long long tileBase = ((long long)blockIdx.x * 8 + wave) * 16;

  // ---- layer 0: A direct from global (K=32) ----
  v16h a0;
  {
    long long r = tileBase + ln;
    bool ok = r < E;
    const float* src = xe + r * 32;
    int k1 = half * 8, k2 = 16 + half * 8;
    #pragma unroll
    for (int j = 0; j < 8; ++j) {
      a0[j]     = ok ? (_Float16)src[k1 + j] : (_Float16)0.f;
      a0[8 + j] = ok ? (_Float16)src[k2 + j] : (_Float16)0.f;
    }
  }
  v8f acc[4];
  #pragma unroll
  for (int ct = 0; ct < 4; ++ct) {
    v8f c = {0.f, 0.f, 0.f, 0.f, 0.f, 0.f, 0.f, 0.f};
    acc[ct] = wmma_f16(a0, load_bfrag(frags, 0 + ct, lane), c);
  }

  relu_stage_next(acc, b0, st, frags, 4, lane);    // -> layer 1
  relu_stage_next(acc, b1, st, frags, 12, lane);   // -> layer 2
  relu_stage_next(acc, b2, st, frags, 20, lane);   // -> layer 3

  // ---- bias + LayerNorm + store + scatter ----
  float bb[4], sc[4], bi[4];
  #pragma unroll
  for (int ct = 0; ct < 4; ++ct) {
    int col = ct * 16 + ln;
    bb[ct] = b3[col]; sc[ct] = lns[col]; bi[ct] = lnb[col];
  }
  #pragma unroll
  for (int r = 0; r < 8; ++r) {
    float w[4]; float s = 0.f, q = 0.f;
    #pragma unroll
    for (int ct = 0; ct < 4; ++ct) { w[ct] = acc[ct][r] + bb[ct]; s += w[ct]; q += w[ct] * w[ct]; }
    #pragma unroll
    for (int m = 1; m <= 8; m <<= 1) { s += __shfl_xor(s, m, 32); q += __shfl_xor(q, m, 32); }
    float mean = s * (1.f / 64.f);
    float var  = q * (1.f / 64.f) - mean * mean;
    float rstd = rsqrtf(var + LN_EPS);
    long long gr = tileBase + half * 8 + r;
    if (gr < E) {
      int d0 = eidx[gr * 2 + 0], d1 = eidx[gr * 2 + 1];
      #pragma unroll
      for (int ct = 0; ct < 4; ++ct) {
        int col = ct * 16 + ln;
        float o = (w[ct] - mean) * rstd * sc[ct] + bi[ct];
        out_edge[gr * HID + col] = o;
        unsafeAtomicAdd(&agg[(long long)d0 * HID + col], o);
        unsafeAtomicAdd(&agg[(long long)d1 * HID + col], o);
      }
    }
  }
}

// Node MLP on [x_node | agg] (K=96) + LayerNorm. agg aliases out_node (read-before-write per row).
__global__ void __launch_bounds__(256)
node_kernel(const float* __restrict__ xn, const _Float16* __restrict__ frags,
            const float* __restrict__ b0, const float* __restrict__ b1,
            const float* __restrict__ b2, const float* __restrict__ b3,
            const float* __restrict__ lns, const float* __restrict__ lnb,
            const float* __restrict__ agg, float* __restrict__ out_node, long long N) {
  __shared__ __align__(16) _Float16 stage[8 * 16 * HID];
  int lane = threadIdx.x & 31, wave = threadIdx.x >> 5;
  int half = lane >> 4, ln = lane & 15;
  _Float16* st = &stage[wave * 16 * HID];
  long long tileBase = ((long long)blockIdx.x * 8 + wave) * 16;

  // ---- layer 0: A from concat(x_node[0:32], agg[0:64]) (K=96, 3 chunks) ----
  v16h a[3];
  {
    long long r = tileBase + ln;
    bool ok = r < N;
    const float* xr = xn + r * 32;
    const float* ar = agg + r * HID;
    #pragma unroll
    for (int kc = 0; kc < 3; ++kc) {
      #pragma unroll
      for (int j = 0; j < 8; ++j) {
        int K1 = kc * 32 + half * 8 + j;
        int K2 = K1 + 16;
        float f1 = ok ? (K1 < 32 ? xr[K1] : ar[K1 - 32]) : 0.f;
        float f2 = ok ? (K2 < 32 ? xr[K2] : ar[K2 - 32]) : 0.f;
        a[kc][j] = (_Float16)f1;
        a[kc][8 + j] = (_Float16)f2;
      }
    }
  }
  v8f acc[4];
  #pragma unroll
  for (int ct = 0; ct < 4; ++ct) {
    v8f c = {0.f, 0.f, 0.f, 0.f, 0.f, 0.f, 0.f, 0.f};
    c = wmma_f16(a[0], load_bfrag(frags, 28 + 0 * 4 + ct, lane), c);
    c = wmma_f16(a[1], load_bfrag(frags, 28 + 1 * 4 + ct, lane), c);
    c = wmma_f16(a[2], load_bfrag(frags, 28 + 2 * 4 + ct, lane), c);
    acc[ct] = c;
  }

  relu_stage_next(acc, b0, st, frags, 40, lane);
  relu_stage_next(acc, b1, st, frags, 48, lane);
  relu_stage_next(acc, b2, st, frags, 56, lane);

  float bb[4], sc[4], bi[4];
  #pragma unroll
  for (int ct = 0; ct < 4; ++ct) {
    int col = ct * 16 + ln;
    bb[ct] = b3[col]; sc[ct] = lns[col]; bi[ct] = lnb[col];
  }
  #pragma unroll
  for (int r = 0; r < 8; ++r) {
    float w[4]; float s = 0.f, q = 0.f;
    #pragma unroll
    for (int ct = 0; ct < 4; ++ct) { w[ct] = acc[ct][r] + bb[ct]; s += w[ct]; q += w[ct] * w[ct]; }
    #pragma unroll
    for (int m = 1; m <= 8; m <<= 1) { s += __shfl_xor(s, m, 32); q += __shfl_xor(q, m, 32); }
    float mean = s * (1.f / 64.f);
    float var  = q * (1.f / 64.f) - mean * mean;
    float rstd = rsqrtf(var + LN_EPS);
    long long gr = tileBase + half * 8 + r;
    if (gr < N) {
      #pragma unroll
      for (int ct = 0; ct < 4; ++ct) {
        int col = ct * 16 + ln;
        out_node[gr * HID + col] = (w[ct] - mean) * rstd * sc[ct] + bi[ct];
      }
    }
  }
}

extern "C" void kernel_launch(void* const* d_in, const int* in_sizes, int n_in,
                              void* d_out, int out_size, void* d_ws, size_t ws_size,
                              hipStream_t stream) {
  (void)n_in; (void)out_size; (void)ws_size;
  const float* x_edge = (const float*)d_in[0];
  const float* x_node = (const float*)d_in[1];
  // mlp0_layers: (W,b) x4 -> d_in[2..9]; ln0 -> 10,11; mlp1_layers -> 12..19; ln1 -> 20,21
  const float* Ws[8] = {(const float*)d_in[2],  (const float*)d_in[4],
                        (const float*)d_in[6],  (const float*)d_in[8],
                        (const float*)d_in[12], (const float*)d_in[14],
                        (const float*)d_in[16], (const float*)d_in[18]};
  const float* b0_0 = (const float*)d_in[3];
  const float* b0_1 = (const float*)d_in[5];
  const float* b0_2 = (const float*)d_in[7];
  const float* b0_3 = (const float*)d_in[9];
  const float* ln0s = (const float*)d_in[10];
  const float* ln0b = (const float*)d_in[11];
  const float* b1_0 = (const float*)d_in[13];
  const float* b1_1 = (const float*)d_in[15];
  const float* b1_2 = (const float*)d_in[17];
  const float* b1_3 = (const float*)d_in[19];
  const float* ln1s = (const float*)d_in[20];
  const float* ln1b = (const float*)d_in[21];
  const int*   eidx = (const int*)d_in[22];

  long long E = (long long)in_sizes[0] / 32;
  long long N = (long long)in_sizes[1] / 32;

  _Float16* frags = (_Float16*)d_ws;          // 64 fragments * 512 halfs = 64 KB
  float* out_node = (float*)d_out;            // [N,64] — doubles as agg accumulator
  float* out_edge = out_node + N * (long long)HID;  // [E,64]
  float* agg      = out_node;

  // Zero the aggregation buffer (= x_node3 output region) every call.
  hipMemsetAsync(agg, 0, (size_t)(N * HID) * sizeof(float), stream);

  // Pack all 8 weight matrices into B-fragment layout (f16).
  static const int kcs[8]   = {1, 2, 2, 2, 3, 2, 2, 2};
  static const int bases[8] = {0, 4, 12, 20, 28, 40, 48, 56};  // fragment index bases
  for (int i = 0; i < 8; ++i) {
    int total = kcs[i] * 2048;
    pack_weights_kernel<<<(total + 255) / 256, 256, 0, stream>>>(
        Ws[i], frags + (size_t)bases[i] * 512, kcs[i]);
  }

  // Edge MLP + LN + scatter-add.
  int eblocks = (int)((E + 127) / 128);
  edge_kernel<<<eblocks, 256, 0, stream>>>(x_edge, frags, b0_0, b0_1, b0_2, b0_3,
                                           ln0s, ln0b, eidx, out_edge, agg, E);

  // Node MLP + LN.
  int nblocks = (int)((N + 127) / 128);
  node_kernel<<<nblocks, 256, 0, stream>>>(x_node, frags, b1_0, b1_1, b1_2, b1_3,
                                           ln1s, ln1b, agg, out_node, N);
}
